// fairDMoN_49220325212394
// MI455X (gfx1250) — matrix operand order repaired and checked by reading, not compiled, not run on hardware
//
#include <hip/hip_runtime.h>

typedef __bf16 bf16_t;
typedef bf16_t v16bf __attribute__((ext_vector_type(16)));
typedef bf16_t v8bf  __attribute__((ext_vector_type(8)));
typedef bf16_t v2bf  __attribute__((ext_vector_type(2)));
typedef float  v8f   __attribute__((ext_vector_type(8)));

#define N_NODES 50000
#define DIM     256
#define KCL     64

// ds_swizzle_b32, group-of-32 mode: and=0x1f, or=0, xor=mask  -> butterfly
#define SWZ_XOR_F(x, mask) \
  __int_as_float(__builtin_amdgcn_ds_swizzle(__float_as_int(x), (0x1f | ((mask) << 10))))

// ---------------------------------------------------------------------------
// ws layout (floats):
//   [0,64)            cluster sizes cs[k]
//   [64,256)          Sd[3][64]
//   [256,259)         tr_pool[3]
//   [259,262)         ne[3]
//   [320, 320+3N)     d_full, d_red, d_blue
//   [150336, +16384)  P[64][256] split-K accumulator
#define WS_ZERO_FLOATS 166720

__global__ __launch_bounds__(256) void k_zero(float* __restrict__ p, int n) {
  for (int i = blockIdx.x * 256 + threadIdx.x; i < n; i += gridDim.x * 256)
    p[i] = 0.0f;
}

// ---------------------------------------------------------------------------
// Kernel 1: S = softmax(F @ W + b) via v_wmma_f32_16x16x32_bf16.
// Block = 256 threads (8 waves); each wave owns a 16x64 output tile.
__global__ __launch_bounds__(256)
void k_gemm_softmax(const float* __restrict__ F, const float* __restrict__ W,
                    const float* __restrict__ bias, float* __restrict__ Sout,
                    float* __restrict__ cs_g) {
  __shared__ bf16_t Wt[64][264];   // W transposed: Wt[n][k], row pitch 16B-aligned
  __shared__ float  csum[64];

  const int tid  = threadIdx.x;
  const int lane = tid & 31;
  const int w    = tid >> 5;

  // stage W transposed as bf16; packed pair stores (ds_store_b32)
  for (int idx = tid; idx < 64 * 128; idx += 256) {
    int n = idx & 63, kp = idx >> 6;           // kp 0..127 -> k = 2kp, 2kp+1
    v2bf t;
    t[0] = (bf16_t)W[(2 * kp) * 64 + n];       // coalesced across n
    t[1] = (bf16_t)W[(2 * kp + 1) * 64 + n];
    *(v2bf*)&Wt[n][2 * kp] = t;
  }
  if (tid < 64) csum[tid] = 0.0f;
  __syncthreads();

  const int m0 = blockIdx.x * 128 + w * 16;
  const int hi = lane >> 4;
  const int nl = lane & 15;
  int mA = m0 + nl; if (mA > N_NODES - 1) mA = N_NODES - 1;   // clamp, EXEC stays full
  const float* __restrict__ Arow = F + (size_t)mA * DIM;

  const int koffA = hi ? 16 : 0;   // A k order chosen here (contiguous, same as B): lanes0-15 K=0..15, hi lanes K=16..31
  const int koffB = hi ? 16 : 0;   // B k order chosen here (contiguous, same as A): lanes0-15 K=0..15, hi lanes K=16..31

  v8f acc[4];
#pragma unroll
  for (int nt = 0; nt < 4; nt++)
#pragma unroll
    for (int j = 0; j < 8; j++) acc[nt][j] = 0.0f;

  for (int kk = 0; kk < DIM; kk += 32) {
    float4 a0 = *(const float4*)(Arow + kk + koffA);
    float4 a1 = *(const float4*)(Arow + kk + koffA + 4);
    float4 a2 = *(const float4*)(Arow + kk + koffA + 8);
    float4 a3 = *(const float4*)(Arow + kk + koffA + 12);
    v16bf a;
    a[0]  = (bf16_t)a0.x; a[1]  = (bf16_t)a0.y; a[2]  = (bf16_t)a0.z; a[3]  = (bf16_t)a0.w;
    a[4]  = (bf16_t)a1.x; a[5]  = (bf16_t)a1.y; a[6]  = (bf16_t)a1.z; a[7]  = (bf16_t)a1.w;
    a[8]  = (bf16_t)a2.x; a[9]  = (bf16_t)a2.y; a[10] = (bf16_t)a2.z; a[11] = (bf16_t)a2.w;
    a[12] = (bf16_t)a3.x; a[13] = (bf16_t)a3.y; a[14] = (bf16_t)a3.z; a[15] = (bf16_t)a3.w;
#pragma unroll
    for (int nt = 0; nt < 4; nt++) {
      int n = nt * 16 + nl;
      const v8bf blo = *(const v8bf*)&Wt[n][kk + koffB];
      const v8bf bhi = *(const v8bf*)&Wt[n][kk + koffB + 8];
      v16bf b;
#pragma unroll
      for (int i = 0; i < 8; i++) { b[i] = blo[i]; b[8 + i] = bhi[i]; }
      acc[nt] = __builtin_amdgcn_wmma_f32_16x16x32_bf16(
          false, a, false, b, (short)0, acc[nt], false, false);
    }
  }

  // bias add (C layout: VGPR j -> row j (+8 for hi lanes), col = nt*16+nl)
#pragma unroll
  for (int nt = 0; nt < 4; nt++) {
    float bl = bias[nt * 16 + nl];
#pragma unroll
    for (int j = 0; j < 8; j++) acc[nt][j] += bl;
  }

  // in-register softmax: row (j, half) spans 16 lanes x 4 accumulators.
  // 16-lane butterfly via ds_swizzle (xor masks stay inside each half).
#pragma unroll
  for (int j = 0; j < 8; j++) {
    float mx = fmaxf(fmaxf(acc[0][j], acc[1][j]), fmaxf(acc[2][j], acc[3][j]));
    mx = fmaxf(mx, SWZ_XOR_F(mx, 8));
    mx = fmaxf(mx, SWZ_XOR_F(mx, 4));
    mx = fmaxf(mx, SWZ_XOR_F(mx, 2));
    mx = fmaxf(mx, SWZ_XOR_F(mx, 1));
    float s = 0.0f;
#pragma unroll
    for (int nt = 0; nt < 4; nt++) {
      float e = __expf(acc[nt][j] - mx); acc[nt][j] = e; s += e;
    }
    s += SWZ_XOR_F(s, 8);
    s += SWZ_XOR_F(s, 4);
    s += SWZ_XOR_F(s, 2);
    s += SWZ_XOR_F(s, 1);
    float inv = 1.0f / s;
#pragma unroll
    for (int nt = 0; nt < 4; nt++) acc[nt][j] *= inv;
  }

  // store S + per-block cluster-size partials
  float csl[4] = {0.f, 0.f, 0.f, 0.f};
#pragma unroll
  for (int j = 0; j < 8; j++) {
    int m = m0 + j + hi * 8;
    if (m < N_NODES) {
#pragma unroll
      for (int nt = 0; nt < 4; nt++) {
        float v = acc[nt][j];
        Sout[(size_t)m * KCL + nt * 16 + nl] = v;
        csl[nt] += v;
      }
    }
  }
#pragma unroll
  for (int nt = 0; nt < 4; nt++) atomicAdd(&csum[nt * 16 + nl], csl[nt]);
  __syncthreads();
  if (tid < 64) atomicAdd(&cs_g[tid], csum[tid]);
}

// ---------------------------------------------------------------------------
// Kernel 2: per-edge  tr += v*<S[r],S[c]>,  d[c] += v,  ne += v.
// Two edges per wave per iteration: lanes 0-15 own edge e0, lanes 16-31 own
// e1; lane nl gathers float4 (cols 4nl..4nl+3). 4-step 16-lane swizzle
// butterfly for the dot; software-pipelined index loads + S-row prefetch.
__global__ __launch_bounds__(256)
void k_edges(const int* __restrict__ er, const int* __restrict__ ec,
             const float* __restrict__ ev, int E, const float* __restrict__ S,
             float* __restrict__ dcol, float* __restrict__ tr_slot,
             float* __restrict__ ne_slot) {
  const int lane = threadIdx.x & 31;
  const int w    = threadIdx.x >> 5;
  const int hi   = lane >> 4;                   // which edge of the pair
  const int nl   = lane & 15;                   // column group
  const long gp  = (long)(blockIdx.x * 8 + w) * 2;
  const long np  = (long)gridDim.x * 8 * 2;

  float tr_acc = 0.0f, ne_acc = 0.0f;

  // prologue: load first pair's indices
  long e = gp + hi;
  bool vcur = (e < E);
  long es = vcur ? e : 0;
  int r = er[es], c = ec[es];
  float v = vcur ? ev[es] : 0.0f;

  for (long e0 = gp; e0 < E; e0 += np) {
    const int rr_ = r, cc_ = c;
    const float vv = v;
    const bool vnow = vcur;

    // pipeline: fetch next pair's indices, prefetch its S rows
    long en = e0 + np + hi;
    vcur = (en < E);
    long esn = vcur ? en : 0;
    r = er[esn]; c = ec[esn];
    v = vcur ? ev[esn] : 0.0f;
    __builtin_prefetch(S + (size_t)r * KCL + nl * 4, 0, 0);   // global_prefetch_b8
    __builtin_prefetch(S + (size_t)c * KCL + nl * 4, 0, 0);

    float4 sr = *(const float4*)(S + (size_t)rr_ * KCL + nl * 4);
    float4 sc = *(const float4*)(S + (size_t)cc_ * KCL + nl * 4);
    float p = sr.x * sc.x + sr.y * sc.y + sr.z * sc.z + sr.w * sc.w;
    p += SWZ_XOR_F(p, 1);
    p += SWZ_XOR_F(p, 2);
    p += SWZ_XOR_F(p, 4);
    p += SWZ_XOR_F(p, 8);
    if (nl == 0) {                       // lanes 0 and 16: one per edge
      tr_acc += vv * p;
      ne_acc += vv;
      if (vnow) atomicAdd(&dcol[cc_], vv);
    }
  }
  if (nl == 0) {
    atomicAdd(tr_slot, tr_acc);
    atomicAdd(ne_slot, ne_acc);
  }
}

// ---------------------------------------------------------------------------
// Kernel 3: Sd[g][k] = sum_n S[n][k] * d_g[n]   (3 graphs at once)
__global__ __launch_bounds__(256)
void k_sd(const float* __restrict__ S, const float* __restrict__ d0,
          const float* __restrict__ d1, const float* __restrict__ d2,
          float* __restrict__ Sd) {
  const int lane = threadIdx.x & 31;
  const int w    = threadIdx.x >> 5;
  const int gw   = blockIdx.x * 8 + w;
  const int nw   = gridDim.x * 8;
  float s0x = 0, s0y = 0, s1x = 0, s1y = 0, s2x = 0, s2y = 0;
  for (int n = gw; n < N_NODES; n += nw) {
    float2 s = *(const float2*)(S + (size_t)n * KCL + lane * 2);
    float a = d0[n], b = d1[n], c = d2[n];
    s0x += s.x * a; s0y += s.y * a;
    s1x += s.x * b; s1y += s.y * b;
    s2x += s.x * c; s2y += s.y * c;
  }
  atomicAdd(&Sd[0 * 64 + lane * 2],     s0x);
  atomicAdd(&Sd[0 * 64 + lane * 2 + 1], s0y);
  atomicAdd(&Sd[1 * 64 + lane * 2],     s1x);
  atomicAdd(&Sd[1 * 64 + lane * 2 + 1], s1y);
  atomicAdd(&Sd[2 * 64 + lane * 2],     s2x);
  atomicAdd(&Sd[2 * 64 + lane * 2 + 1], s2y);
}

// ---------------------------------------------------------------------------
// Kernel 4: split-K WMMA GEMM  P[64][256] = S^T @ F.
// Per 32-row chunk: stage S^T and F^T as bf16 in LDS (transposed so both
// WMMA fragments load contiguous 16B runs), accumulate, atomic-add at end.
__global__ __launch_bounds__(256)
void k_pool_gemm(const float* __restrict__ S, const float* __restrict__ F,
                 float* __restrict__ P) {
  __shared__ bf16_t StL[64][40];    // StL[k][n_local], pitch 80B (16B aligned)
  __shared__ bf16_t FtL[256][40];   // FtL[d][n_local]

  const int tid  = threadIdx.x;
  const int lane = tid & 31;
  const int w    = tid >> 5;
  const int hi   = lane >> 4;
  const int nl   = lane & 15;
  const int mt   = w >> 1;          // 0..3  -> cluster rows mt*16..mt*16+15
  const int half = w & 1;           // 0..1  -> feature cols half*128..+127
  const int koffA = hi ? 16 : 0;
  const int koffB = hi ? 16 : 0;
  const int mrow = mt * 16 + nl;

  v8f acc[8];
#pragma unroll
  for (int j = 0; j < 8; j++)
#pragma unroll
    for (int i = 0; i < 8; i++) acc[j][i] = 0.0f;

  const int numChunks = (N_NODES + 31) / 32;
  for (int ch = blockIdx.x; ch < numChunks; ch += gridDim.x) {
    const int nb = ch * 32;
    __syncthreads();
    // stage S^T chunk, packed pair stores
    for (int idx = tid; idx < 64 * 16; idx += 256) {
      int k = idx & 63, rp = idx >> 6;
      int n0 = nb + 2 * rp, n1 = n0 + 1;
      v2bf t;
      t[0] = (bf16_t)((n0 < N_NODES) ? S[(size_t)n0 * KCL + k] : 0.0f);
      t[1] = (bf16_t)((n1 < N_NODES) ? S[(size_t)n1 * KCL + k] : 0.0f);
      *(v2bf*)&StL[k][2 * rp] = t;
    }
    // stage features^T chunk, packed pair stores
    for (int idx = tid; idx < 256 * 16; idx += 256) {
      int d = idx & 255, rp = idx >> 8;
      int n0 = nb + 2 * rp, n1 = n0 + 1;
      v2bf t;
      t[0] = (bf16_t)((n0 < N_NODES) ? F[(size_t)n0 * DIM + d] : 0.0f);
      t[1] = (bf16_t)((n1 < N_NODES) ? F[(size_t)n1 * DIM + d] : 0.0f);
      *(v2bf*)&FtL[d][2 * rp] = t;
    }
    __syncthreads();

    const v8bf alo = *(const v8bf*)&StL[mrow][koffA];
    const v8bf ahi = *(const v8bf*)&StL[mrow][koffA + 8];
    v16bf a;
#pragma unroll
    for (int i = 0; i < 8; i++) { a[i] = alo[i]; a[8 + i] = ahi[i]; }
#pragma unroll
    for (int j = 0; j < 8; j++) {
      int dcol = half * 128 + j * 16 + nl;
      const v8bf blo = *(const v8bf*)&FtL[dcol][koffB];
      const v8bf bhi = *(const v8bf*)&FtL[dcol][koffB + 8];
      v16bf b;
#pragma unroll
      for (int i = 0; i < 8; i++) { b[i] = blo[i]; b[8 + i] = bhi[i]; }
      acc[j] = __builtin_amdgcn_wmma_f32_16x16x32_bf16(
          false, a, false, b, (short)0, acc[j], false, false);
    }
  }

#pragma unroll
  for (int j = 0; j < 8; j++) {
    int dcol = half * 128 + j * 16 + nl;
#pragma unroll
    for (int i = 0; i < 8; i++) {
      int m = mt * 16 + i + hi * 8;
      atomicAdd(&P[m * DIM + dcol], acc[j][i]);
    }
  }
}

// ---------------------------------------------------------------------------
// Kernel 5: pooled = selu(P / cs)  +  scalar total loss.
__global__ __launch_bounds__(256)
void k_final(const float* __restrict__ P, const float* __restrict__ cs,
             const float* __restrict__ Sd, const float* __restrict__ trp,
             const float* __restrict__ ne, const float* __restrict__ lam,
             float* __restrict__ pooled, float* __restrict__ lossOut) {
  int idx = blockIdx.x * 256 + threadIdx.x;
  if (idx < KCL * DIM) {
    int m = idx >> 8;
    float x = P[idx] / cs[m];
    const float scale = 1.0507009873554805f;
    const float alpha = 1.6732632423543772f;
    pooled[idx] = (x > 0.0f) ? scale * x : scale * alpha * (__expf(x) - 1.0f);
  }
  if (blockIdx.x == 0 && threadIdx.x == 0) {
    float two_m = ne[0];
    float L[3];
    for (int g = 0; g < 3; g++) {
      float sd2 = 0.0f;
      for (int k = 0; k < 64; k++) { float v = Sd[g * 64 + k]; sd2 += v * v; }
      float trn = sd2 * 0.5f / ne[g];
      L[g] = -(trp[g] - trn) * 0.5f / two_m;
    }
    float fair = fabsf(lam[0] * (L[1] - L[2]));
    float cs2 = 0.0f;
    for (int k = 0; k < 64; k++) cs2 += cs[k] * cs[k];
    float collapse = sqrtf(cs2) / (float)N_NODES * 8.0f - 1.0f;  // sqrt(K)=8
    lossOut[0] = fair + L[0] + 0.1f * collapse;
  }
}

// ---------------------------------------------------------------------------
extern "C" void kernel_launch(void* const* d_in, const int* in_sizes, int n_in,
                              void* d_out, int out_size, void* d_ws, size_t ws_size,
                              hipStream_t stream) {
  const float* F    = (const float*)d_in[0];
  const int*   rows = (const int*)d_in[1];
  const int*   cols = (const int*)d_in[2];
  const float* vals = (const float*)d_in[3];
  const int*   rr   = (const int*)d_in[4];
  const int*   rc   = (const int*)d_in[5];
  const float* rv   = (const float*)d_in[6];
  const int*   br   = (const int*)d_in[7];
  const int*   bc   = (const int*)d_in[8];
  const float* bv   = (const float*)d_in[9];
  const float* W    = (const float*)d_in[10];
  const float* bias = (const float*)d_in[11];
  const float* lam  = (const float*)d_in[12];

  float* out    = (float*)d_out;
  float* pooled = out;                                   // [64*256]
  float* Sout   = out + 16384;                           // [N*64]
  float* loss   = out + 16384 + (size_t)N_NODES * KCL;   // [1]

  float* wsf   = (float*)d_ws;
  float* cs    = wsf;
  float* Sd    = wsf + 64;
  float* trp   = wsf + 256;
  float* ne    = wsf + 259;
  float* dfull = wsf + 320;
  float* dred  = dfull + N_NODES;
  float* dblue = dred + N_NODES;
  float* P     = wsf + 150336;

  const int E   = in_sizes[1];
  const int E2r = in_sizes[4];
  const int E2b = in_sizes[7];

  k_zero<<<652, 256, 0, stream>>>(wsf, WS_ZERO_FLOATS);
  k_gemm_softmax<<<(N_NODES + 127) / 128, 256, 0, stream>>>(F, W, bias, Sout, cs);
  k_edges<<<512, 256, 0, stream>>>(rows, cols, vals, E,   Sout, dfull, trp + 0, ne + 0);
  k_edges<<<512, 256, 0, stream>>>(rr,   rc,   rv,   E2r, Sout, dred,  trp + 1, ne + 1);
  k_edges<<<512, 256, 0, stream>>>(br,   bc,   bv,   E2b, Sout, dblue, trp + 2, ne + 2);
  k_sd<<<256, 256, 0, stream>>>(Sout, dfull, dred, dblue, Sd);
  k_pool_gemm<<<128, 256, 0, stream>>>(Sout, F, P);
  k_final<<<64, 256, 0, stream>>>(P, cs, Sd, trp, ne, lam, pooled, loss);
}
